// SimpleDecoder_4629974745236
// MI455X (gfx1250) — matrix operand order repaired and checked by reading, not compiled, not run on hardware
//
#include <hip/hip_runtime.h>
#include <math.h>

// ---------------------------------------------------------------------------
// MI455X (gfx1250) decoder block, bf16 WMMA everywhere.
//   q = x@Wq + bq ; attn = tril(q qT)/8 @ q ; y1 = attn@Wo + bo + x
//   h = LN1(y1)   ; up = gelu(h@Wu + bu)    ; y2 = up@Wd + bd + h ; out = LN2(y2)
// mask input is all-ones -> tril applied analytically.
// GEMM staging uses CDNA5 GLOBAL_LOAD_ASYNC_TO_LDS_B128 (ASYNCcnt path).
// ---------------------------------------------------------------------------

typedef unsigned short u16;
typedef unsigned int   u32;
typedef unsigned long long u64;
typedef __attribute__((ext_vector_type(16))) __bf16 v16bf;
typedef __attribute__((ext_vector_type(8)))  float  v8f;
typedef __attribute__((ext_vector_type(4)))  u32    vu4;   // POD 16-byte vector
typedef __attribute__((ext_vector_type(4)))  float  vf4;

#define BDIM    1024
#define SEQ     1024
#define BATCH   8
#define HEADS_N 16
#define HD      64
#define MROWS   (BATCH * SEQ)   // 8192
#define D4      (4 * BDIM)      // 4096

constexpr int EPI_BF16  = 0;   // outB = bf16(acc + bias)
constexpr int EPI_RESID = 1;   // outF = acc + bias + resid
constexpr int EPI_GELU  = 2;   // outB = bf16(gelu_exact(acc + bias))

// fp32 -> bf16, round-to-nearest-even (pure integer, no __bf16 arithmetic)
__device__ __forceinline__ u16 f2bf(float f) {
  u32 u = __float_as_uint(f);
  return (u16)((u + 0x7FFFu + ((u >> 16) & 1u)) >> 16);
}

union FragU { v16bf v; vu4 q[2]; };

// A-operand fragment (16x32 bf16): lane-half h holds K = kb+8h..+7 and kb+16+8h..+7
__device__ __forceinline__ v16bf frag_a(const u16* row, int kb, int h) {
  FragU u;
  u.q[0] = *(const vu4*)(row + kb + 8 * h);
  u.q[1] = *(const vu4*)(row + kb + 16 + 8 * h);
  return u.v;
}
// B-operand fragment (32x16 bf16): lane-half h holds K = kb+8h..+7 and kb+16+8h..+7
__device__ __forceinline__ v16bf frag_b(const u16* row, int kb, int h) {
  FragU u;
  u.q[0] = *(const vu4*)(row + kb + 8 * h);
  u.q[1] = *(const vu4*)(row + kb + 16 + 8 * h);
  return u.v;
}

__device__ __forceinline__ v8f wmma_bf16(v16bf a, v16bf b, v8f c) {
  // D = A*B + C ; 16x16 f32 accumulator, 8 VGPRs
  return __builtin_amdgcn_wmma_f32_16x16x32_bf16(false, a, false, b,
                                                 (short)0, c, false, false);
}

// CDNA5 async global->LDS copy, 16B per lane (GVS mode: saddr + 32-bit voffset).
// lds_off: per-lane LDS byte address (low 32 bits of generic __shared__ ptr).
__device__ __forceinline__ void async_load_b128(u32 lds_off, u64 saddr, u32 voff) {
  asm volatile("global_load_async_to_lds_b128 %0, %1, %2"
               :: "v"(lds_off), "v"(voff), "s"(saddr) : "memory");
}
__device__ __forceinline__ void wait_async0() {
  asm volatile("s_wait_asynccnt 0" ::: "memory");
}

// ---------------------------------------------------------------------------
// fp32 -> bf16 cast (x)
// ---------------------------------------------------------------------------
__global__ __launch_bounds__(256) void cast_bf16(const float* __restrict__ x,
                                                 u16* __restrict__ xb, int n) {
  int i = (blockIdx.x * 256 + threadIdx.x) * 4;
  if (i < n) {
    vf4 v = *(const vf4*)(x + i);
    xb[i + 0] = f2bf(v[0]); xb[i + 1] = f2bf(v[1]);
    xb[i + 2] = f2bf(v[2]); xb[i + 3] = f2bf(v[3]);
  }
}

// ---------------------------------------------------------------------------
// W[K,N] f32 -> Wt[N,K] bf16 (so GEMM B-operand rows are K-contiguous)
// ---------------------------------------------------------------------------
__global__ __launch_bounds__(256) void transpose_cast(const float* __restrict__ W,
                                                      u16* __restrict__ Wt,
                                                      int K, int N) {
  __shared__ float tile[32][33];
  int k0 = blockIdx.y * 32, n0 = blockIdx.x * 32;
  int tx = threadIdx.x & 31, ty = threadIdx.x >> 5;  // 32 x 8
#pragma unroll
  for (int i = 0; i < 4; i++) {
    int k = ty + i * 8;
    tile[k][tx] = W[(size_t)(k0 + k) * N + n0 + tx];
  }
  __syncthreads();
#pragma unroll
  for (int i = 0; i < 4; i++) {
    int n = ty + i * 8;
    Wt[(size_t)(n0 + n) * K + k0 + tx] = f2bf(tile[tx][n]);
  }
}

// ---------------------------------------------------------------------------
// Tiled WMMA GEMM: C[M,N] = A[M,K](bf16) @ Bt[N,K]^T(bf16) + epilogue
// 128x128 block tile, BK=64, 8 waves, each wave 2x4 WMMA tiles x 2 K-steps.
// Staging via GLOBAL_LOAD_ASYNC_TO_LDS_B128 (no VGPR round-trip).
// ---------------------------------------------------------------------------
template <int EPI>
__global__ __launch_bounds__(256)
void gemm_bf16(const u16* __restrict__ A, const u16* __restrict__ Bt,
               const float* __restrict__ bias, const float* __restrict__ resid,
               float* __restrict__ outF, u16* __restrict__ outB,
               int M, int N, int K) {
  constexpr int BK = 64;
  constexpr int LDT = 80;  // 64 + 16 pad -> 160B rows (16B aligned, spread banks)
  __shared__ u16 As[128 * LDT];   // 20 KB
  __shared__ u16 Bs[128 * LDT];   // 20 KB

  const int tid = threadIdx.x;
  const int lane = tid & 31, wid = tid >> 5;
  const int wm = wid & 3, wn = wid >> 2;       // 4x2 wave grid -> 32 rows x 64 cols
  const int half = lane >> 4, mr = lane & 15;
  const int bm0 = blockIdx.y * 128, bn0 = blockIdx.x * 128;
  const int lr = tid >> 2;                     // 0..63 staging row
  const int lc = (tid & 3) * 8;                // 0,8,16,24 staging col (+32)

  const u64 aAddr = (u64)(uintptr_t)A;
  const u64 bAddr = (u64)(uintptr_t)Bt;

  v8f zero = {};
  v8f acc[2][4];
#pragma unroll
  for (int i = 0; i < 2; i++)
#pragma unroll
    for (int j = 0; j < 4; j++) acc[i][j] = zero;

  for (int kb = 0; kb < K; kb += BK) {
    // async-stage A and Bt 128x64 tiles: 4 x b128 per thread per matrix
#pragma unroll
    for (int p = 0; p < 128; p += 64) {
      int row = p + lr;
#pragma unroll
      for (int c = 0; c < 64; c += 32) {
        int col = lc + c;
        u32 aoff = (u32)(((size_t)(bm0 + row) * K + kb + col) * 2);
        u32 boff = (u32)(((size_t)(bn0 + row) * K + kb + col) * 2);
        async_load_b128((u32)(uintptr_t)&As[row * LDT + col], aAddr, aoff);
        async_load_b128((u32)(uintptr_t)&Bs[row * LDT + col], bAddr, boff);
      }
    }
    if (kb + BK < K) {  // global_prefetch of next K-slab
      __builtin_prefetch(A + (size_t)(bm0 + lr) * K + kb + BK + lc, 0, 3);
      __builtin_prefetch(Bt + (size_t)(bn0 + lr) * K + kb + BK + lc, 0, 3);
    }
    wait_async0();
    __syncthreads();

#pragma unroll
    for (int kk = 0; kk < BK; kk += 32) {
      v16bf af[2], bf[4];
#pragma unroll
      for (int i = 0; i < 2; i++)
        af[i] = frag_a(&As[(wm * 32 + i * 16 + mr) * LDT], kk, half);
#pragma unroll
      for (int j = 0; j < 4; j++)
        bf[j] = frag_b(&Bs[(wn * 64 + j * 16 + mr) * LDT], kk, half);
#pragma unroll
      for (int i = 0; i < 2; i++)
#pragma unroll
        for (int j = 0; j < 4; j++) acc[i][j] = wmma_bf16(af[i], bf[j], acc[i][j]);
    }
    __syncthreads();
  }

  // epilogue: C-layout row = 8*half + r, col = lane%16
#pragma unroll
  for (int i = 0; i < 2; i++) {
#pragma unroll
    for (int j = 0; j < 4; j++) {
#pragma unroll
      for (int r = 0; r < 8; r++) {
        int row = bm0 + wm * 32 + i * 16 + 8 * half + r;
        int col = bn0 + wn * 64 + j * 16 + mr;
        float v = acc[i][j][r] + bias[col];
        size_t idx = (size_t)row * N + col;
        if constexpr (EPI == EPI_BF16) {
          outB[idx] = f2bf(v);
        } else if constexpr (EPI == EPI_RESID) {
          outF[idx] = v + resid[idx];
        } else {  // exact-erf GELU (torch nn.GELU)
          float g = 0.5f * v * (1.0f + erff(v * 0.70710678118654752f));
          outB[idx] = f2bf(g);
        }
      }
    }
  }
}

// ---------------------------------------------------------------------------
// Fused causal attention (no softmax, masked scores zeroed, k=v=q):
//   attn[b,h,q,:] = sum_{k<=q} (q_q . q_k / 8) * q_k
// One block per (b,h, 64-query tile); key blocks of 64 streamed through LDS.
// Score tensor (537 MB) never touches HBM.
// ---------------------------------------------------------------------------
__global__ __launch_bounds__(256)
void attention_kernel(const u16* __restrict__ qb, u16* __restrict__ attnb) {
  constexpr int LDT = 80;  // 64 + 16 pad -> 160B rows (16B aligned)
  __shared__ u16 qt[64 * LDT];  // queries, row-major [q][hd]
  __shared__ u16 kv[64 * LDT];  // keys,    row-major [k][hd]   (B of QK^T)
  __shared__ u16 vt[64 * LDT];  // values,  transposed [hd][k]  (B of S*V)
  __shared__ u16 st[64 * LDT];  // bf16 masked scores [q][k]    (A of S*V)

  const int tid = threadIdx.x;
  const int lane = tid & 31, wid = tid >> 5;
  const int wm = wid & 3, wn = wid >> 2;
  const int half = lane >> 4, mr = lane & 15;

  const int qi = blockIdx.x;
  const int b = blockIdx.y / HEADS_N;
  const int h = blockIdx.y % HEADS_N;
  const int q0 = qi * 64;
  const u16* base = qb + (size_t)b * SEQ * BDIM + h * HD;  // row s stride = BDIM

  const int lr = tid >> 2;        // 0..63
  const int lc = (tid & 3) * 8;   // 0,8,16,24  (+32 for second half)

  {  // stage query tile once
    const u16* src = base + (size_t)(q0 + lr) * BDIM;
    *(vu4*)(&qt[lr * LDT + lc]) = *(const vu4*)(src + lc);
    *(vu4*)(&qt[lr * LDT + lc + 32]) = *(const vu4*)(src + lc + 32);
  }

  v8f zero = {};
  v8f oacc[2] = {zero, zero};

  for (int kblk = 0; kblk <= qi; ++kblk) {
    const int k0 = kblk * 64;
    __syncthreads();  // previous iter's kv/vt/st reads done
    {                 // stage keys row-major + values transposed
      const u16* src = base + (size_t)(k0 + lr) * BDIM;
      vu4 d0 = *(const vu4*)(src + lc);
      vu4 d1 = *(const vu4*)(src + lc + 32);
      *(vu4*)(&kv[lr * LDT + lc]) = d0;
      *(vu4*)(&kv[lr * LDT + lc + 32]) = d1;
      union { vu4 q; u16 e[8]; } u0, u1;
      u0.q = d0; u1.q = d1;
#pragma unroll
      for (int e = 0; e < 8; e++) {
        vt[(lc + e) * LDT + lr] = u0.e[e];
        vt[(lc + 32 + e) * LDT + lr] = u1.e[e];
      }
    }
    __syncthreads();

    // S = q @ k^T  (K-dim = hd = 64 -> two WMMA steps)
    v8f sacc[2] = {zero, zero};
#pragma unroll
    for (int kk = 0; kk < 64; kk += 32) {
      v16bf a = frag_a(&qt[(wm * 16 + mr) * LDT], kk, half);
#pragma unroll
      for (int j = 0; j < 2; j++) {
        v16bf bb = frag_b(&kv[((wn * 2 + j) * 16 + mr) * LDT], kk, half);
        sacc[j] = wmma_bf16(a, bb, sacc[j]);
      }
    }
    // scale by 1/sqrt(64), causal-zero, stage bf16 scores
#pragma unroll
    for (int j = 0; j < 2; j++) {
      int kcol = (wn * 2 + j) * 16 + mr;
      int kidx = k0 + kcol;
#pragma unroll
      for (int r = 0; r < 8; r++) {
        int qrow = wm * 16 + 8 * half + r;
        float v = (kidx <= q0 + qrow) ? sacc[j][r] * 0.125f : 0.0f;
        st[qrow * LDT + kcol] = f2bf(v);
      }
    }
    __syncthreads();

    // attn += S @ V  (K-dim = 64 keys -> two WMMA steps)
#pragma unroll
    for (int kk = 0; kk < 64; kk += 32) {
      v16bf a = frag_a(&st[(wm * 16 + mr) * LDT], kk, half);
#pragma unroll
      for (int j = 0; j < 2; j++) {
        v16bf bb = frag_b(&vt[((wn * 2 + j) * 16 + mr) * LDT], kk, half);
        oacc[j] = wmma_bf16(a, bb, oacc[j]);
      }
    }
  }

  // write [B,H,S,hd] back interleaved as [M, D] bf16
#pragma unroll
  for (int j = 0; j < 2; j++) {
    int col = h * HD + (wn * 2 + j) * 16 + mr;
#pragma unroll
    for (int r = 0; r < 8; r++) {
      int row = q0 + wm * 16 + 8 * half + r;
      attnb[(size_t)(b * SEQ + row) * BDIM + col] = f2bf(oacc[j][r]);
    }
  }
}

// ---------------------------------------------------------------------------
// LayerNorm over D=1024 (block per row). Biased variance, eps inside sqrt.
// ---------------------------------------------------------------------------
__global__ __launch_bounds__(256)
void layernorm_kernel(const float* __restrict__ y, const float* __restrict__ g,
                      const float* __restrict__ bta, float* __restrict__ out,
                      u16* __restrict__ outb) {
  __shared__ float rs[256], rq[256];
  const int row = blockIdx.x;
  const float* p = y + (size_t)row * BDIM;
  float s = 0.f, q = 0.f, vals[4];
#pragma unroll
  for (int i = 0; i < 4; i++) {
    float v = p[threadIdx.x + i * 256];
    vals[i] = v; s += v; q += v * v;
  }
  rs[threadIdx.x] = s; rq[threadIdx.x] = q;
  __syncthreads();
  for (int off = 128; off > 0; off >>= 1) {
    if (threadIdx.x < off) {
      rs[threadIdx.x] += rs[threadIdx.x + off];
      rq[threadIdx.x] += rq[threadIdx.x + off];
    }
    __syncthreads();
  }
  float mean = rs[0] * (1.0f / BDIM);
  float var = rq[0] * (1.0f / BDIM) - mean * mean;
  float rstd = rsqrtf(var + 1e-5f);
#pragma unroll
  for (int i = 0; i < 4; i++) {
    int c = threadIdx.x + i * 256;
    float o = (vals[i] - mean) * rstd * g[c] + bta[c];
    out[(size_t)row * BDIM + c] = o;
    if (outb) outb[(size_t)row * BDIM + c] = f2bf(o);
  }
}

// ---------------------------------------------------------------------------
extern "C" void kernel_launch(void* const* d_in, const int* in_sizes, int n_in,
                              void* d_out, int out_size, void* d_ws, size_t ws_size,
                              hipStream_t stream) {
  const float* x  = (const float*)d_in[0];
  // d_in[1]: mask (all ones; tril applied analytically in attention_kernel)
  const float* Wq = (const float*)d_in[2];
  const float* bq = (const float*)d_in[3];
  const float* Wo = (const float*)d_in[4];
  const float* bo = (const float*)d_in[5];
  const float* Wu = (const float*)d_in[6];
  const float* bu = (const float*)d_in[7];
  const float* Wd = (const float*)d_in[8];
  const float* bd = (const float*)d_in[9];
  const float* g1 = (const float*)d_in[10];
  const float* b1 = (const float*)d_in[11];
  const float* g2 = (const float*)d_in[12];
  const float* b2 = (const float*)d_in[13];
  float* out = (float*)d_out;

  const size_t MD = (size_t)MROWS * BDIM;   // 8.39M elements
  const size_t MD4 = (size_t)MROWS * D4;
  char* w = (char*)d_ws;                    // ~255 MB total
  u16* xb    = (u16*)w;   w += MD * 2;
  u16* qbuf  = (u16*)w;   w += MD * 2;
  u16* attnb = (u16*)w;   w += MD * 2;
  u16* hb    = (u16*)w;   w += MD * 2;
  u16* upb   = (u16*)w;   w += MD4 * 2;
  u16* WqT   = (u16*)w;   w += (size_t)BDIM * BDIM * 2;
  u16* WoT   = (u16*)w;   w += (size_t)BDIM * BDIM * 2;
  u16* WuT   = (u16*)w;   w += (size_t)D4 * BDIM * 2;
  u16* WdT   = (u16*)w;   w += (size_t)BDIM * D4 * 2;
  float* y1  = (float*)w; w += MD * 4;
  float* hf  = (float*)w; w += MD * 4;
  float* y2  = (float*)w; w += MD * 4;

  // 1) one-time casts / weight transposes (bf16)
  cast_bf16<<<dim3((u32)(MD / 4 / 256)), 256, 0, stream>>>(x, xb, (int)MD);
  transpose_cast<<<dim3(BDIM / 32, BDIM / 32), 256, 0, stream>>>(Wq, WqT, BDIM, BDIM);
  transpose_cast<<<dim3(BDIM / 32, BDIM / 32), 256, 0, stream>>>(Wo, WoT, BDIM, BDIM);
  transpose_cast<<<dim3(D4 / 32, BDIM / 32), 256, 0, stream>>>(Wu, WuT, BDIM, D4);
  transpose_cast<<<dim3(BDIM / 32, D4 / 32), 256, 0, stream>>>(Wd, WdT, D4, BDIM);

  // 2) q = x@Wq + bq (bf16 out; q==k==v)
  gemm_bf16<EPI_BF16><<<dim3(BDIM / 128, MROWS / 128), 256, 0, stream>>>(
      xb, WqT, bq, nullptr, nullptr, qbuf, MROWS, BDIM, BDIM);

  // 3) fused causal attention
  attention_kernel<<<dim3(SEQ / 64, BATCH * HEADS_N), 256, 0, stream>>>(qbuf, attnb);

  // 4) y1 = attn@Wo + bo + x
  gemm_bf16<EPI_RESID><<<dim3(BDIM / 128, MROWS / 128), 256, 0, stream>>>(
      attnb, WoT, bo, x, y1, nullptr, MROWS, BDIM, BDIM);

  // 5) h = LN1(y1)  (f32 for final residual + bf16 for next GEMM)
  layernorm_kernel<<<dim3(MROWS), 256, 0, stream>>>(y1, g1, b1, hf, hb);

  // 6) up = gelu(h@Wu + bu) (bf16 out)
  gemm_bf16<EPI_GELU><<<dim3(D4 / 128, MROWS / 128), 256, 0, stream>>>(
      hb, WuT, bu, nullptr, nullptr, upb, MROWS, D4, BDIM);

  // 7) y2 = up@Wd + bd + h
  gemm_bf16<EPI_RESID><<<dim3(BDIM / 128, MROWS / 128), 256, 0, stream>>>(
      upb, WdT, bd, hf, y2, nullptr, MROWS, BDIM, D4);

  // 8) out = LN2(y2)
  layernorm_kernel<<<dim3(MROWS), 256, 0, stream>>>(y2, g2, b2, out, nullptr);
}
